// Delan_Sin_28243704939038
// MI455X (gfx1250) — matrix operand order repaired and checked by reading, not compiled, not run on hardware
//
#include <hip/hip_runtime.h>

// DeLaN-style network, algebraically fused (exact identities):
//   stage1: pre[b,0:90] = q @ [ld_w1;lo_w1;g_w1]^T + bias            (K=7)
//   stage2: z_m = [sin_ld,sin_lo,qDDot] @ U_m^T + ub_m   (30x67, precomputed)
//           z_c = [cos_ld,cos_lo,qDot ] @ U_c^T + c_b1   (30x67, precomputed:
//                 U_c[j,h] = sum_{o,d} c_w1[j,o*7+d] * W2[o,h] * W1[h,d])
//           g   = sin_g @ g_w2^T + g_b2                  (7x30)
//   stage3: out = sig(z_m)@m_w2^T + sig(z_c)@c_w2^T + g + (m_b2+c_b2)
// All GEMMs mapped to v_wmma_f32_16x16x32_f16, batch on the M dimension.
// Weight fragments live in block-shared LDS (keeps VGPRs < 256, no MSB thrash).

#define WPB 4  // waves per block (wave32)

typedef __attribute__((ext_vector_type(16))) _Float16 v16h;
typedef __attribute__((ext_vector_type(8)))  float    v8f;

struct __attribute__((aligned(16))) U4 { unsigned x, y, z, w; };

union Frag {
  unsigned u[8];
  U4       q[2];
  v16h     h;
};

__device__ __forceinline__ unsigned pk2(float a, float b) {
  union { _Float16 h[2]; unsigned u; } t;
  t.h[0] = (_Float16)a; t.h[1] = (_Float16)b;
  return t.u;
}

__device__ __forceinline__ v8f wmma_f16(v16h a, v16h b, v8f c) {
  // (neg_a, A, neg_b, B, c_mod, C, reuse_a, reuse_b)
  return __builtin_amdgcn_wmma_f32_16x16x32_f16(false, a, false, b, (short)0, c,
                                                false, false);
}

__device__ __forceinline__ v8f splat8(float b) {
  v8f c;
#pragma unroll
  for (int i = 0; i < 8; i++) c[i] = b;
  return c;
}

__device__ __forceinline__ float sigmoid_fast(float x) {
  return __builtin_amdgcn_rcpf(1.0f + __expf(-x));
}

// A-fragment (16x32 f16) gather from per-wave LDS feature rows.
// k order used here (same as B pack): lanes 0-15 hold M=0..15 / K = 0..15 (v0..7);
//           lanes 16-31 hold K = 16..31.
__device__ __forceinline__ v16h load_a_lds(const _Float16* rowbase, int strideH,
                                           int kstep, int lane) {
  const int mm = lane & 15, gg = lane >> 4;
  const _Float16* p = rowbase + mm * strideH + kstep * 32 + 16 * gg;
  Frag u;
  u.q[0] = *(const U4*)(p);        // K = base .. base+7   (4 dwords)
  u.q[1] = *(const U4*)(p + 8);    // K = base+8 .. +15    (4 dwords)
  return u.h;
}

// B-fragment fetch from block-shared LDS copy of the packed weights.
__device__ __forceinline__ v16h load_b_lds(const unsigned* fragL, int f, int lane) {
  const U4* p = (const U4*)(fragL + (f * 32 + lane) * 8);
  Frag u;
  u.q[0] = p[0];
  u.q[1] = p[1];
  return u.h;
}

// ---------------------------------------------------------------------------
// Pack kernel: build fused matrices directly in WMMA B-fragment layout.
// B k order used here (32x16 f16, matches A gather): lanes 0-15 hold K=0..15 (pairs per VGPR),
//                       lanes 16-31 hold K=16..31; N = lane & 15.
// ---------------------------------------------------------------------------
__device__ float bval(int stage, int k, int n,
                      const float* ld_w1, const float* ld_w2,
                      const float* lo_w1, const float* lo_w2,
                      const float* g_w1,  const float* g_w2,
                      const float* m_w1,  const float* m_w2,
                      const float* c_w1,  const float* c_w2) {
  float s = 0.f;
  switch (stage) {
    case 0:  // [ld_w1;lo_w1;g_w1]^T, 90 outputs, K<7
      if (k < 7 && n < 90) {
        if (n < 30)      s = ld_w1[n * 7 + k];
        else if (n < 60) s = lo_w1[(n - 30) * 7 + k];
        else             s = g_w1[(n - 60) * 7 + k];
      }
      break;
    case 1:  // U_m : K = [sin_ld(30) | sin_lo(30) | qDDot(7)]
      if (n < 30) {
        if (k < 30)      { for (int o = 0; o < 7;  o++) s += m_w1[n*35 + o]     * ld_w2[o*30 + k]; }
        else if (k < 60) { for (int o = 0; o < 21; o++) s += m_w1[n*35 + 7 + o] * lo_w2[o*30 + (k-30)]; }
        else if (k < 67) s = m_w1[n*35 + 28 + (k - 60)];
      }
      break;
    case 2:  // U_c : K = [cos_ld(30) | cos_lo(30) | qDot(7)]
      if (n < 30) {
        if (k < 30) {
          for (int o = 0; o < 7; o++)
            for (int d = 0; d < 7; d++)
              s += c_w1[n*203 + o*7 + d] * ld_w2[o*30 + k] * ld_w1[k*7 + d];
        } else if (k < 60) {
          const int kk = k - 30;
          for (int o = 0; o < 21; o++)
            for (int d = 0; d < 7; d++)
              s += c_w1[n*203 + (7 + o)*7 + d] * lo_w2[o*30 + kk] * lo_w1[kk*7 + d];
        } else if (k < 67) s = c_w1[n*203 + 196 + (k - 60)];
      }
      break;
    case 3:  // g_w2 (7x30)
      if (n < 7 && k < 30) s = g_w2[n*30 + k];
      break;
    case 4:  // [m_w2 | c_w2] : K = [sig_m(30) | sig_c(30)]
      if (n < 7) {
        if (k < 30)      s = m_w2[n*30 + k];
        else if (k < 60) s = c_w2[n*30 + (k - 30)];
      }
      break;
  }
  return s;
}

__global__ void delan_pack(const float* ld_w1, const float* ld_b1, const float* ld_w2, const float* ld_b2,
                           const float* lo_w1, const float* lo_b1, const float* lo_w2, const float* lo_b2,
                           const float* g_w1,  const float* g_b1,  const float* g_w2,  const float* g_b2,
                           const float* m_w1,  const float* m_b1,  const float* m_w2,  const float* m_b2,
                           const float* c_w1,  const float* c_b1,  const float* c_w2,  const float* c_b2,
                           unsigned* frag, float* bias) {
  const int tid = threadIdx.x + blockIdx.x * blockDim.x;
  const int stride = blockDim.x * gridDim.x;
  // 21 fragments x 32 lanes x 8 dwords
  for (int i = tid; i < 21 * 32 * 8; i += stride) {
    const int j = i & 7, l = (i >> 3) & 31, f = i >> 8;
    int stage, ntile = 0, kstep = 0;
    if (f < 6)        { stage = 0; ntile = f; }
    else if (f < 12)  { stage = 1; ntile = (f - 6) / 3;  kstep = (f - 6) % 3; }
    else if (f < 18)  { stage = 2; ntile = (f - 12) / 3; kstep = (f - 12) % 3; }
    else if (f == 18) { stage = 3; }
    else              { stage = 4; kstep = f - 19; }
    const int n  = ntile * 16 + (l & 15);
    const int k0 = kstep * 32 + (l >> 4) * 16 + 2 * j;
    const float v0 = bval(stage, k0,     n, ld_w1, ld_w2, lo_w1, lo_w2, g_w1, g_w2, m_w1, m_w2, c_w1, c_w2);
    const float v1 = bval(stage, k0 + 1, n, ld_w1, ld_w2, lo_w1, lo_w2, g_w1, g_w2, m_w1, m_w2, c_w1, c_w2);
    frag[i] = pk2(v0, v1);
  }
  // 12 C-tile bias vectors of 16 (seed values for accumulators)
  for (int i = tid; i < 192; i += stride) {
    const int ct = i >> 4, nn = i & 15;
    float v = 0.f;
    if (ct < 6) {                       // stage-1 tiles: [ld_b1;lo_b1;g_b1]
      const int n = ct * 16 + nn;
      if (n < 30)      v = ld_b1[n];
      else if (n < 60) v = lo_b1[n - 30];
      else if (n < 90) v = g_b1[n - 60];
    } else if (ct < 8) {                // z_m bias: m_b1 + m_w1[:, :28] @ [ld_b2;lo_b2]
      const int n = (ct - 6) * 16 + nn;
      if (n < 30) {
        v = m_b1[n];
        for (int o = 0; o < 7;  o++) v += m_w1[n*35 + o]     * ld_b2[o];
        for (int o = 0; o < 21; o++) v += m_w1[n*35 + 7 + o] * lo_b2[o];
      }
    } else if (ct < 10) {               // z_c bias
      const int n = (ct - 8) * 16 + nn;
      if (n < 30) v = c_b1[n];
    } else if (ct == 10) {              // g bias
      if (nn < 7) v = g_b2[nn];
    } else {                            // final bias
      if (nn < 7) v = m_b2[nn] + c_b2[nn];
    }
    bias[i] = v;
  }
}

// ---------------------------------------------------------------------------
// Main kernel: one 16-sample tile per wave per iteration. Weight fragments
// in block-shared LDS; transcendental hand-off via wave-private LDS scratch.
// ---------------------------------------------------------------------------
__global__ void __launch_bounds__(WPB * 32)
delan_main(const float* __restrict__ x, const unsigned* __restrict__ frag,
           const float* __restrict__ bias, float* __restrict__ out, int ntiles) {
  // [ fragments: 5376 dwords | per-wave scratch: WPB x 2304 dwords ]
  __shared__ __attribute__((aligned(16))) unsigned shm[5376 + WPB * 2304];
  const int lane = threadIdx.x & 31;
  const int wave = threadIdx.x >> 5;

  unsigned* fragL = shm;
  _Float16* wsc = (_Float16*)(shm + 5376) + wave * 4608;
  _Float16* f2a = wsc;             // 16 x 96 : [sin_ld | sin_lo | qDDot | 0]
  _Float16* f2b = wsc + 16 * 96;   // 16 x 96 : [cos_ld | cos_lo | qDot  | 0]
  _Float16* f2g = f2b + 16 * 96;   // 16 x 32 : [sin_g | 0]
  _Float16* f3  = f2g + 16 * 32;   // 16 x 64 : [sig_m | sig_c | 0]

  // zero per-wave scratch once: pad regions are never rewritten, data regions
  // are fully rewritten every iteration (all traffic wave-private after init).
  { unsigned* z = (unsigned*)wsc;
    for (int i = lane; i < 2304; i += 32) z[i] = 0u; }

  // cooperative copy of packed weight fragments into LDS (1344 x 16B)
  for (int i = threadIdx.x; i < 1344; i += WPB * 32)
    ((U4*)fragL)[i] = ((const U4*)frag)[i];
  __syncthreads();

  float bb[12];
#pragma unroll
  for (int t = 0; t < 12; t++) bb[t] = bias[t * 16 + (lane & 15)];

  const int m = lane & 15;
  const int g = lane >> 4;
  const int gwave   = blockIdx.x * WPB + wave;
  const int wstride = gridDim.x * WPB;

  for (int tile = gwave; tile < ntiles; tile += wstride) {
    const long base = (long)tile * 16;
    if (tile + wstride < ntiles)
      __builtin_prefetch(x + ((long)(tile + wstride) * 16 + m) * 21, 0, 0);

    // build A1 (q in K=0..6 of a 16x32 tile) and stash qDot/qDDot features
    unsigned d0 = 0, d1 = 0, d2 = 0, d3 = 0;
    if (lane < 16) {
      const float* xr = x + (base + m) * 21;
      float q[7], qd[7], qdd[7];
#pragma unroll
      for (int d = 0; d < 7; d++) { q[d] = xr[d]; qd[d] = xr[7 + d]; qdd[d] = xr[14 + d]; }
      d0 = pk2(q[0], q[1]); d1 = pk2(q[2], q[3]);
      d2 = pk2(q[4], q[5]); d3 = pk2(q[6], 0.f);
#pragma unroll
      for (int d = 0; d < 7; d++) {
        f2a[m * 96 + 60 + d] = (_Float16)qdd[d];
        f2b[m * 96 + 60 + d] = (_Float16)qd[d];
      }
    }
    Frag ua;
    ua.u[0] = d0; ua.u[1] = d1; ua.u[2] = d2; ua.u[3] = d3;
    ua.u[4] = 0;  ua.u[5] = 0;  ua.u[6] = 0;  ua.u[7] = 0;
    const v16h a1 = ua.h;

    // ---- stage 1: pre = q @ Wcat^T + b  (6 N-tiles), then sin/cos to LDS
#pragma unroll
    for (int ct = 0; ct < 6; ct++) {
      v8f c = splat8(bb[ct]);
      c = wmma_f16(a1, load_b_lds(fragL, ct, lane), c);
      const int n = ct * 16 + m;  // C layout: N = lane&15, M = vgpr + 8*(lane>>4)
#pragma unroll
      for (int v = 0; v < 8; v++) {
        const float pre = c[v];
        const int mm = v + 8 * g;
        if (n < 60) {
          f2a[mm * 96 + n] = (_Float16)__sinf(pre);
          f2b[mm * 96 + n] = (_Float16)__cosf(pre);
        } else if (n < 90) {
          f2g[mm * 32 + (n - 60)] = (_Float16)__sinf(pre);
        }
      }
    }

    // ---- stage 2: z_m = [sin_ld,sin_lo,qDDot] @ U_m^T  (2 N-tiles x 3 K-steps)
#pragma unroll
    for (int t = 0; t < 2; t++) {
      v8f c = splat8(bb[6 + t]);
#pragma unroll
      for (int s = 0; s < 3; s++)
        c = wmma_f16(load_a_lds(f2a, 96, s, lane),
                     load_b_lds(fragL, 6 + t * 3 + s, lane), c);
      const int n = t * 16 + m;
      if (n < 30) {
#pragma unroll
        for (int v = 0; v < 8; v++)
          f3[(v + 8 * g) * 64 + n] = (_Float16)sigmoid_fast(c[v]);
      }
    }
    // ---- stage 2: z_c = [cos_ld,cos_lo,qDot] @ U_c^T
#pragma unroll
    for (int t = 0; t < 2; t++) {
      v8f c = splat8(bb[8 + t]);
#pragma unroll
      for (int s = 0; s < 3; s++)
        c = wmma_f16(load_a_lds(f2b, 96, s, lane),
                     load_b_lds(fragL, 12 + t * 3 + s, lane), c);
      const int n = t * 16 + m;
      if (n < 30) {
#pragma unroll
        for (int v = 0; v < 8; v++)
          f3[(v + 8 * g) * 64 + 30 + n] = (_Float16)sigmoid_fast(c[v]);
      }
    }
    // ---- stage 2: g head (stays in registers; seeds stage 3 accumulator)
    v8f cg = splat8(bb[10]);
    cg = wmma_f16(load_a_lds(f2g, 32, 0, lane), load_b_lds(fragL, 18, lane), cg);

    // ---- stage 3: out = [sig_m,sig_c] @ [m_w2|c_w2]^T + g + bias
    v8f c3;
#pragma unroll
    for (int v = 0; v < 8; v++) c3[v] = cg[v] + bb[11];
#pragma unroll
    for (int s = 0; s < 2; s++)
      c3 = wmma_f16(load_a_lds(f3, 64, s, lane), load_b_lds(fragL, 19 + s, lane), c3);

    if (m < 7) {
#pragma unroll
      for (int v = 0; v < 8; v++)
        out[(base + v + 8 * g) * 7 + m] = c3[v];
    }
  }
}

extern "C" void kernel_launch(void* const* d_in, const int* in_sizes, int n_in,
                              void* d_out, int out_size, void* d_ws, size_t ws_size,
                              hipStream_t stream) {
  const float* x     = (const float*)d_in[0];
  const float* ld_w1 = (const float*)d_in[1];
  const float* ld_b1 = (const float*)d_in[2];
  const float* ld_w2 = (const float*)d_in[3];
  const float* ld_b2 = (const float*)d_in[4];
  const float* lo_w1 = (const float*)d_in[5];
  const float* lo_b1 = (const float*)d_in[6];
  const float* lo_w2 = (const float*)d_in[7];
  const float* lo_b2 = (const float*)d_in[8];
  const float* g_w1  = (const float*)d_in[9];
  const float* g_b1  = (const float*)d_in[10];
  const float* g_w2  = (const float*)d_in[11];
  const float* g_b2  = (const float*)d_in[12];
  const float* m_w1  = (const float*)d_in[13];
  const float* m_b1  = (const float*)d_in[14];
  const float* m_w2  = (const float*)d_in[15];
  const float* m_b2  = (const float*)d_in[16];
  const float* c_w1  = (const float*)d_in[17];
  const float* c_b1  = (const float*)d_in[18];
  const float* c_w2  = (const float*)d_in[19];
  const float* c_b2  = (const float*)d_in[20];

  unsigned* frag = (unsigned*)d_ws;                               // 21*32*8 dwords
  float* bias = (float*)((char*)d_ws + 21 * 32 * 8 * sizeof(unsigned));  // 192 floats

  const int batch  = in_sizes[0] / 21;
  const int ntiles = batch / 16;

  delan_pack<<<dim3(8), dim3(256), 0, stream>>>(
      ld_w1, ld_b1, ld_w2, ld_b2, lo_w1, lo_b1, lo_w2, lo_b2,
      g_w1, g_b1, g_w2, g_b2, m_w1, m_b1, m_w2, m_b2,
      c_w1, c_b1, c_w2, c_b2, frag, bias);

  int blocks = (ntiles + WPB - 1) / WPB;
  if (blocks > 16384) blocks = 16384;
  if (blocks < 1) blocks = 1;
  delan_main<<<dim3(blocks), dim3(WPB * 32), 0, stream>>>(
      x, frag, bias, (float*)d_out, ntiles);
}
